// MambaEncoder_8615704396480
// MI455X (gfx1250) — compile-verified, matrix operand order checked by reading, not run on hardware
//
#include <hip/hip_runtime.h>
#include <hip/hip_bf16.h>
#include <math.h>

typedef __attribute__((ext_vector_type(16))) __bf16 v16bf;
typedef __attribute__((ext_vector_type(8)))  float  v8f;
typedef long long ll;

#if defined(__has_builtin)
#if __has_builtin(__builtin_amdgcn_tensor_load_to_lds)
#define HAVE_TDM 1
#endif
#endif

#ifdef HAVE_TDM
typedef __attribute__((ext_vector_type(4))) unsigned int u32x4;
typedef __attribute__((ext_vector_type(8))) int          i32x8;
typedef __attribute__((ext_vector_type(4))) int          i32x4;

// Issue a TDM load of a 2-D f32 tile (32 cols x 64 rows, row stride = rs elems)
// into LDS at lds_off. LDS rows are padded 32+4 dwords -> matches float[64][36].
// tensor_dim0/1 = remaining cols/rows so the TDM zero-fills tile edges.
__device__ __forceinline__ void tdm_load_tile_f32(const float* lds_dst, const float* gptr,
                                                  int rows_avail, int cols_avail, int rs)
{
    unsigned lds_off = (unsigned)(unsigned long long)lds_dst;   // LDS aperture: low 32b = offset
    unsigned long long ga = (unsigned long long)(const void*)gptr;
    u32x4 g0;
    g0.x = 1u;                                       // count=1 (valid user D#)
    g0.y = lds_off;                                  // lds_addr
    g0.z = (unsigned)(ga & 0xFFFFFFFFull);           // global_addr[31:0]
    g0.w = (unsigned)((ga >> 32) & 0x01FFFFFFu) | (2u << 30);  // addr[56:32], type=2
    unsigned td0 = (unsigned)cols_avail;
    unsigned td1 = (unsigned)rows_avail;
    i32x8 g1;
    // data_size=4B (17:16), pad_enable (20), pad_interval=32dw (24:22=4), pad_amount=4dw (31:25=3)
    g1[0] = (int)((2u << 16) | (1u << 20) | (4u << 22) | (3u << 25));
    g1[1] = (int)((td0 & 0xFFFFu) << 16);                              // tensor_dim0[15:0]
    g1[2] = (int)(((td0 >> 16) & 0xFFFFu) | ((td1 & 0xFFFFu) << 16));  // dim0[31:16], dim1[15:0]
    g1[3] = (int)(((td1 >> 16) & 0xFFFFu) | (32u << 16));              // dim1[31:16], tile_dim0=32
    g1[4] = 64;                                                        // tile_dim1=64, tile_dim2=0
    g1[5] = (int)(unsigned)rs;                                         // tensor_dim0_stride[31:0]
    g1[6] = 0;
    g1[7] = 0;
    i32x4 z4 = {0, 0, 0, 0};
#if __has_include(<hip/amd_detail/amd_gfx1250_TDM.h>)
    i32x8 z8 = {0, 0, 0, 0, 0, 0, 0, 0};
    __builtin_amdgcn_tensor_load_to_lds(g0, g1, z4, z4, z8, 0);   // clang-23 / therock arity
#else
    __builtin_amdgcn_tensor_load_to_lds(g0, g1, z4, z4, 0);       // ROCm 7.2 arity
#endif
}
#endif // HAVE_TDM

// ---- manual staging helpers (branch-free edge handling: clamp + select) ----

// Stage rows [tile0, tile0+64) of a [R x K] row-major matrix, K-slice k0..k0+32.
__device__ __forceinline__ void stage_rows_kcontig(const float* __restrict__ Src,
                                                   float (*Dst)[36],
                                                   int tile0, int R, int K, int k0, int tid)
{
    const int row = tid >> 1;
    const int cb  = (tid & 1) << 4;
    const int gr  = tile0 + row;
    if ((tile0 + 64 <= R) && (k0 + 32 <= K)) {
        const float* src = Src + (ll)gr * K + k0 + cb;
        float v[16];
        #pragma unroll
        for (int i = 0; i < 16; ++i) v[i] = src[i];
        #pragma unroll
        for (int i = 0; i < 16; ++i) Dst[row][cb + i] = v[i];
        if (k0 + 32 < K)
            __builtin_prefetch(src + 32, 0, 1);   // global_prefetch_b8
    } else {
        const int grc = gr < R ? gr : R - 1;
        #pragma unroll
        for (int i = 0; i < 16; ++i) {
            int gk  = k0 + cb + i;
            int gkc = gk < K ? gk : K - 1;
            float v = Src[(ll)grc * K + gkc];            // always in-bounds
            Dst[row][cb + i] = (gr < R && gk < K) ? v : 0.f;
        }
    }
}

// Stage B[K,N] row-major transposed into LDS [n][k] (for C = A*B).
__device__ __forceinline__ void stage_b_nn(const float* __restrict__ B,
                                           float (*Dst)[36],
                                           int tileN, int N, int K, int k0, int tid)
{
    const int krow = tid >> 2;        // 0..31
    const int nb   = (tid & 3) << 4;  // 0,16,32,48
    const int gk   = k0 + krow;
    if ((tileN + 64 <= N) && (k0 + 32 <= K)) {
        const float* src = B + (ll)gk * N + tileN + nb;
        float v[16];
        #pragma unroll
        for (int i = 0; i < 16; ++i) v[i] = src[i];
        #pragma unroll
        for (int i = 0; i < 16; ++i) Dst[nb + i][krow] = v[i];
        if (k0 + 32 < K)
            __builtin_prefetch(src + 32LL * N, 0, 1);
    } else {
        const int gkc = gk < K ? gk : K - 1;
        #pragma unroll
        for (int i = 0; i < 16; ++i) {
            int gn  = tileN + nb + i;
            int gnc = gn < N ? gn : N - 1;
            float v = B[(ll)gkc * N + gnc];              // always in-bounds
            Dst[nb + i][krow] = (gk < K && gn < N) ? v : 0.f;
        }
    }
}

// One 64x64x32 tile: 4 WMMAs per wave from f32 LDS tiles (cvt to bf16 at load).
// CDNA5 16-bit A/B frag layout (wave32): lanes 0-15 hold K {0..7,16..23},
// lanes 16-31 hold K {8..15,24..31}; M/N = lane&15.
__device__ __forceinline__ void gemm_tile(const float (*As)[36], const float (*Bs)[36],
                                          int wRow, int wCol, int fm, int kb,
                                          v8f acc[2][2])
{
    v16bf aF[2], bF[2];
    #pragma unroll
    for (int t2 = 0; t2 < 2; ++t2) {
        const int mr = wRow + t2 * 16 + fm;
        const int nr = wCol + t2 * 16 + fm;
        #pragma unroll
        for (int e = 0; e < 8; ++e) {
            aF[t2][e]     = (__bf16)As[mr][kb + e];
            aF[t2][8 + e] = (__bf16)As[mr][kb + 16 + e];
            bF[t2][e]     = (__bf16)Bs[nr][kb + e];
            bF[t2][8 + e] = (__bf16)Bs[nr][kb + 16 + e];
        }
    }
    #pragma unroll
    for (int i2 = 0; i2 < 2; ++i2)
        #pragma unroll
        for (int j2 = 0; j2 < 2; ++j2)
            acc[i2][j2] = __builtin_amdgcn_wmma_f32_16x16x32_bf16(
                false, aF[i2], false, bF[j2], (short)0, acc[i2][j2], false, false);
}

// ---------------------------------------------------------------------------
// WMMA GEMM:  C[M,N] = A[M,K] * op(B)
//   A row-major (lda=K); BNT=true: B[N,K] (C=A*B^T); BNT=false: B[K,N] (C=A*B)
// Batched over blockIdx.z. NT flavor: both operands DMA'd by the Tensor Data
// Mover, double-buffered (DMA tile kt+1 while WMMA consumes kt, synchronized
// with s_wait_tensorcnt). NN flavor: A via TDM, B transposed manually in LDS.
// ---------------------------------------------------------------------------
template<bool BNT>
__global__ __launch_bounds__(128)
void wmma_gemm_kernel(const float* __restrict__ A, const float* __restrict__ B,
                      float* __restrict__ C, int M, int N, int K,
                      ll sA, ll sB, ll sC)
{
    A += (ll)blockIdx.z * sA;
    B += (ll)blockIdx.z * sB;
    C += (ll)blockIdx.z * sC;

    const int tid   = threadIdx.x;
    const int lane  = tid & 31;
    const int wave  = tid >> 5;
    const int wRow  = (wave >> 1) * 32;
    const int wCol  = (wave & 1) * 32;
    const int tileM = blockIdx.y * 64;
    const int tileN = blockIdx.x * 64;

    v8f acc[2][2] = {};
    const int fm = lane & 15;
    const int kb = (lane >> 4) << 3;
    const int nK = (K + 31) >> 5;

#ifdef HAVE_TDM
    if constexpr (BNT) {
        // -------- fully async path: both tiles via TDM, double-buffered --------
        __shared__ float Asf[2][64][36];
        __shared__ float Bsf[2][64][36];
        if (tid == 0) {
            tdm_load_tile_f32(&Asf[0][0][0], A + (ll)tileM * K, M - tileM, K, K);
            tdm_load_tile_f32(&Bsf[0][0][0], B + (ll)tileN * K, N - tileN, K, K);
        }
        for (int kt = 0; kt < nK; ++kt) {
            const int cur = kt & 1, nxt = cur ^ 1;
            const int k1  = (kt + 1) << 5;
            if (tid == 0 && kt + 1 < nK) {
                tdm_load_tile_f32(&Asf[nxt][0][0], A + (ll)tileM * K + k1, M - tileM, K - k1, K);
                tdm_load_tile_f32(&Bsf[nxt][0][0], B + (ll)tileN * K + k1, N - tileN, K - k1, K);
            }
            if (kt + 1 < nK) __builtin_amdgcn_s_wait_tensorcnt(2);  // current tile landed
            else             __builtin_amdgcn_s_wait_tensorcnt(0);
            __syncthreads();
            gemm_tile(Asf[cur], Bsf[cur], wRow, wCol, fm, kb, acc);
            __syncthreads();
        }
    } else {
        // -------- A via TDM, B transposed manually --------
        __shared__ float Asf[64][36];
        __shared__ float Bsf[64][36];
        for (int kt = 0; kt < nK; ++kt) {
            const int k0 = kt << 5;
            if (tid == 0)
                tdm_load_tile_f32(&Asf[0][0], A + (ll)tileM * K + k0, M - tileM, K - k0, K);
            stage_b_nn(B, Bsf, tileN, N, K, k0, tid);
            __builtin_amdgcn_s_wait_tensorcnt(0);
            __syncthreads();
            gemm_tile(Asf, Bsf, wRow, wCol, fm, kb, acc);
            __syncthreads();
        }
    }
#else
    {
        __shared__ float Asf[64][36];
        __shared__ float Bsf[64][36];
        for (int kt = 0; kt < nK; ++kt) {
            const int k0 = kt << 5;
            stage_rows_kcontig(A, Asf, tileM, M, K, k0, tid);
            if (BNT) stage_rows_kcontig(B, Bsf, tileN, N, K, k0, tid);
            else     stage_b_nn(B, Bsf, tileN, N, K, k0, tid);
            __syncthreads();
            gemm_tile(Asf, Bsf, wRow, wCol, fm, kb, acc);
            __syncthreads();
        }
    }
#endif

    // ---- store D: VGPR v -> M = v + 8*(lane>>4), N = lane&15 ----
    const int cn   = lane & 15;
    const int half = lane >> 4;
    #pragma unroll
    for (int i2 = 0; i2 < 2; ++i2)
        #pragma unroll
        for (int j2 = 0; j2 < 2; ++j2)
            #pragma unroll
            for (int v = 0; v < 8; ++v) {
                int r = tileM + wRow + i2 * 16 + half * 8 + v;
                int c = tileN + wCol + j2 * 16 + cn;
                if (r < M && c < N) C[(ll)r * N + c] = acc[i2][j2][v];
            }
}

// ---------------------------------------------------------------------------
// Direct 3D convolution (cubic volumes). y[co,z,h,w], x[ci,z,h,w].
// ---------------------------------------------------------------------------
__global__ void conv3d_kernel(const float* __restrict__ x, const float* __restrict__ w,
                              const float* __restrict__ b, float* __restrict__ y,
                              int Ci, int Si, int Co, int So, int ks, int stride, int pad,
                              ll total)
{
    ll idx = (ll)blockIdx.x * blockDim.x + threadIdx.x;
    if (idx >= total) return;
    int wo = (int)(idx % So); ll t = idx / So;
    int ho = (int)(t % So);   t /= So;
    int zo = (int)(t % So);
    int co = (int)(t / So);
    float s = b[co];
    for (int ci = 0; ci < Ci; ++ci)
        for (int kz = 0; kz < ks; ++kz) {
            int z = zo * stride - pad + kz;
            if (z < 0 || z >= Si) continue;
            for (int ky = 0; ky < ks; ++ky) {
                int h = ho * stride - pad + ky;
                if (h < 0 || h >= Si) continue;
                for (int kx = 0; kx < ks; ++kx) {
                    int ww = wo * stride - pad + kx;
                    if (ww < 0 || ww >= Si) continue;
                    s += x[(((ll)ci * Si + z) * Si + h) * Si + ww] *
                         w[((((ll)co * Ci + ci) * ks + kz) * ks + ky) * ks + kx];
                }
            }
        }
    y[idx] = s;
}

// Instance norm over spatial dims; one block per channel.
__global__ void instnorm_kernel(const float* __restrict__ x, float* __restrict__ y, ll S)
{
    const int c = blockIdx.x;
    const float* xp = x + (ll)c * S;
    __shared__ float s1[256], s2[256];
    float a = 0.f, b = 0.f;
    for (ll i = threadIdx.x; i < S; i += blockDim.x) { float v = xp[i]; a += v; b += v * v; }
    s1[threadIdx.x] = a; s2[threadIdx.x] = b;
    __syncthreads();
    for (int st = 128; st > 0; st >>= 1) {
        if ((int)threadIdx.x < st) { s1[threadIdx.x] += s1[threadIdx.x + st]; s2[threadIdx.x] += s2[threadIdx.x + st]; }
        __syncthreads();
    }
    float mu  = s1[0] / (float)S;
    float var = s2[0] / (float)S - mu * mu;
    float inv = rsqrtf(var + 1e-5f);
    float* yp = y + (ll)c * S;
    for (ll i = threadIdx.x; i < S; i += blockDim.x) yp[i] = (xp[i] - mu) * inv;
}

// LayerNorm over channel dim; one thread per token. xf: (T, C) in-place.
__global__ void layernorm_kernel(float* __restrict__ xf, const float* __restrict__ w,
                                 const float* __restrict__ b, ll T, int C)
{
    ll t = (ll)blockIdx.x * blockDim.x + threadIdx.x;
    if (t >= T) return;
    float* row = xf + t * C;
    float mu = 0.f;
    for (int c = 0; c < C; ++c) mu += row[c];
    mu /= (float)C;
    float var = 0.f;
    for (int c = 0; c < C; ++c) { float d = row[c] - mu; var += d * d; }
    var /= (float)C;
    float inv = rsqrtf(var + 1e-5f);
    for (int c = 0; c < C; ++c) row[c] = (row[c] - mu) * inv * w[c] + b[c];
}

// tri_pixel gather: x(C,S,S,S) -> xf(p^3, L, C), L=(S/p)^3
__global__ void pix_gather_kernel(const float* __restrict__ x, float* __restrict__ xf,
                                  int C, int S, int p, ll total)
{
    ll idx = (ll)blockIdx.x * blockDim.x + threadIdx.x;
    if (idx >= total) return;
    int sp = S / p, L = sp * sp * sp;
    int c = (int)(idx % C); ll t = idx / C;
    int l = (int)(t % L);
    int bn = (int)(t / L);
    int pz = bn / (p * p), ph = (bn / p) % p, pw = bn % p;
    int z0 = l / (sp * sp), h0 = (l / sp) % sp, w0 = l % sp;
    xf[idx] = x[(((ll)c * S + z0 * p + pz) * S + h0 * p + ph) * S + w0 * p + pw];
}

// tri_pixel scatter+residual: x += out, out(Bn, C, L)
__global__ void pix_scatter_kernel(float* __restrict__ x, const float* __restrict__ out,
                                   int C, int S, int p, ll total)
{
    ll idx = (ll)blockIdx.x * blockDim.x + threadIdx.x;
    if (idx >= total) return;
    int sp = S / p, L = sp * sp * sp;
    int l = (int)(idx % L); ll t = idx / L;
    int c = (int)(t % C);
    int bn = (int)(t / C);
    int pz = bn / (p * p), ph = (bn / p) % p, pw = bn % p;
    int z0 = l / (sp * sp), h0 = (l / sp) % sp, w0 = l % sp;
    x[(((ll)c * S + z0 * p + pz) * S + h0 * p + ph) * S + w0 * p + pw] += out[idx];
}

// Average pool by p: x(C,S,S,S) -> y(C,(S/p)^3)
__global__ void avgpool_kernel(const float* __restrict__ x, float* __restrict__ y,
                               int C, int S, int p, ll total)
{
    ll idx = (ll)blockIdx.x * blockDim.x + threadIdx.x;
    if (idx >= total) return;
    int sp = S / p;
    int w0 = (int)(idx % sp); ll t = idx / sp;
    int h0 = (int)(t % sp);   t /= sp;
    int z0 = (int)(t % sp);
    int c  = (int)(t / sp);
    float s = 0.f;
    for (int dz = 0; dz < p; ++dz)
        for (int dh = 0; dh < p; ++dh)
            for (int dw = 0; dw < p; ++dw)
                s += x[(((ll)c * S + z0 * p + dz) * S + h0 * p + dh) * S + w0 * p + dw];
    y[idx] = s / (float)(p * p * p);
}

// (C,L) -> (L,C)
__global__ void transpose_cl_lc_kernel(const float* __restrict__ x, float* __restrict__ y,
                                       int C, ll L, ll total)
{
    ll idx = (ll)blockIdx.x * blockDim.x + threadIdx.x;
    if (idx >= total) return;
    ll l = idx % L;
    int c = (int)(idx / L);
    y[l * C + c] = x[idx];
}

// Trilinear upsample (half-pixel centers) of small(C,sp,sp,sp) by p, added into x.
__global__ void trilinear_add_kernel(float* __restrict__ x, const float* __restrict__ small,
                                     int C, int S, int p, ll total)
{
    ll idx = (ll)blockIdx.x * blockDim.x + threadIdx.x;
    if (idx >= total) return;
    int sp = S / p;
    int w = (int)(idx % S); ll t = idx / S;
    int h = (int)(t % S);   t /= S;
    int z = (int)(t % S);
    int c = (int)(t / S);
    float fz = (z + 0.5f) / p - 0.5f, fh = (h + 0.5f) / p - 0.5f, fw = (w + 0.5f) / p - 0.5f;
    int z0 = (int)floorf(fz), h0 = (int)floorf(fh), w0 = (int)floorf(fw);
    float az = fz - z0, ah = fh - h0, aw = fw - w0;
    int z0c = min(max(z0, 0), sp - 1), z1c = min(max(z0 + 1, 0), sp - 1);
    int h0c = min(max(h0, 0), sp - 1), h1c = min(max(h0 + 1, 0), sp - 1);
    int w0c = min(max(w0, 0), sp - 1), w1c = min(max(w0 + 1, 0), sp - 1);
    const float* sm = small + (ll)c * sp * sp * sp;
    auto G = [&](int zz, int hh, int ww) { return sm[((ll)zz * sp + hh) * sp + ww]; };
    float v =
        (1.f - az) * ((1.f - ah) * ((1.f - aw) * G(z0c, h0c, w0c) + aw * G(z0c, h0c, w1c)) +
                      ah * ((1.f - aw) * G(z0c, h1c, w0c) + aw * G(z0c, h1c, w1c))) +
        az * ((1.f - ah) * ((1.f - aw) * G(z1c, h0c, w0c) + aw * G(z1c, h0c, w1c)) +
              ah * ((1.f - aw) * G(z1c, h1c, w0c) + aw * G(z1c, h1c, w1c)));
    x[idx] += v;
}

// Sequence permute of xz(Bn,E,L): mode 1=reverse, 2=slice-interleave
__global__ void seq_permute_kernel(const float* __restrict__ xz, float* __restrict__ out,
                                   int L, int ns, int mode, ll total)
{
    ll idx = (ll)blockIdx.x * blockDim.x + threadIdx.x;
    if (idx >= total) return;
    int l = (int)(idx % L);
    ll base = idx - l;
    int src;
    if (mode == 1) src = L - 1 - l;
    else { int Ls = L / ns; src = (l % ns) * Ls + l / ns; }
    out[idx] = xz[base + src];
}

// Depthwise causal conv1d (k=4) + SiLU: xz(Bn,2di,L)[:, :di] -> xc(Bn,di,L)
__global__ void dwconv_silu_kernel(const float* __restrict__ xz, const float* __restrict__ w,
                                   const float* __restrict__ b, float* __restrict__ xc,
                                   int E2, int di, int L, ll total)
{
    ll idx = (ll)blockIdx.x * blockDim.x + threadIdx.x;
    if (idx >= total) return;
    int l = (int)(idx % L); ll t = idx / L;
    int d = (int)(t % di);
    ll bn = t / di;
    const float* row = xz + (bn * E2 + d) * (ll)L;
    float s = b[d];
    #pragma unroll
    for (int j = 0; j < 4; ++j) {
        int li = l - 3 + j;
        if (li >= 0) s += row[li] * w[d * 4 + j];
    }
    xc[idx] = s / (1.f + __expf(-s));
}

// delta = softplus(delta + bias[d]); delta(Bn,di,L)
__global__ void bias_softplus_kernel(float* __restrict__ delta, const float* __restrict__ b,
                                     int di, int L, ll total)
{
    ll idx = (ll)blockIdx.x * blockDim.x + threadIdx.x;
    if (idx >= total) return;
    int d = (int)((idx / L) % di);
    float v = delta[idx] + b[d];
    delta[idx] = (v > 20.f) ? v : log1pf(__expf(v));
}

// Selective scan; one thread per (batch, channel), 16-state recurrence in VGPRs.
__global__ void scan_kernel(const float* __restrict__ delta, const float* __restrict__ u,
                            const float* __restrict__ xdbl, const float* __restrict__ A_log,
                            const float* __restrict__ Dp, const float* xzd,
                            float* __restrict__ y, int Bn, int di, int L, int r, int E2)
{
    int idx = blockIdx.x * blockDim.x + threadIdx.x;
    if (idx >= Bn * di) return;
    int d = idx % di, bn = idx / di;
    float A[16];
    #pragma unroll
    for (int n = 0; n < 16; ++n) A[n] = -__expf(A_log[d * 16 + n]);
    const float Dv = Dp[d];
    const float* dl = delta + ((ll)bn * di + d) * L;
    const float* ul = u     + ((ll)bn * di + d) * L;
    const float* Bl = xdbl  + ((ll)bn * (r + 32) + r) * L;
    const float* Cl = Bl + 16LL * L;
    const float* zl = xzd   + ((ll)bn * E2 + di + d) * L;
    float* yl = y + ((ll)bn * di + d) * L;
    float h[16];
    #pragma unroll
    for (int n = 0; n < 16; ++n) h[n] = 0.f;
    for (int t = 0; t < L; ++t) {
        float dt = dl[t], ut = ul[t];
        float du = dt * ut;
        float acc = 0.f;
        #pragma unroll
        for (int n = 0; n < 16; ++n) {
            h[n] = __expf(dt * A[n]) * h[n] + du * Bl[(ll)n * L + t];
            acc  = fmaf(h[n], Cl[(ll)n * L + t], acc);
        }
        float yv = acc + Dv * ut;
        float z  = zl[t];
        yl[t] = yv * (z / (1.f + __expf(-z)));
    }
}

// acc(Bn,di,L) (+)= y[perm_out(l)]; mode 0=id, 1=reverse, 2=un-slice
__global__ void accum_out_kernel(float* __restrict__ acc, const float* __restrict__ y,
                                 int L, int ns, int mode, int overwrite, ll total)
{
    ll idx = (ll)blockIdx.x * blockDim.x + threadIdx.x;
    if (idx >= total) return;
    int l = (int)(idx % L);
    ll base = idx - l;
    int src;
    if (mode == 0)      src = l;
    else if (mode == 1) src = L - 1 - l;
    else { int Ls = L / ns; src = (l % Ls) * ns + l / Ls; }
    float v = y[base + src];
    acc[idx] = overwrite ? v : acc[idx] + v;
}

// ---------------------------------------------------------------------------
// Host orchestration
// ---------------------------------------------------------------------------
struct DirP  { const float *A_log, *D, *conv_b, *conv_w, *dtproj_b, *dtproj_w, *xproj_w; };
struct LayerP{ DirP b, f, s; const float *in_proj_w, *ln_b, *ln_w, *out_proj_w; };
struct Bufs  { float *XF, *XZ, *XZD, *XC, *XDBL, *DELTA, *YB, *ACC, *OUT; };

static inline int cdiv(int a, int b) { return (a + b - 1) / b; }
static inline int gb(ll n) { return (int)((n + 255) / 256); }

static DirP readDir(void* const* d, int& i) {
    DirP p;
    p.A_log    = (const float*)d[i++]; p.D        = (const float*)d[i++];
    p.conv_b   = (const float*)d[i++]; p.conv_w   = (const float*)d[i++];
    p.dtproj_b = (const float*)d[i++]; p.dtproj_w = (const float*)d[i++];
    p.xproj_w  = (const float*)d[i++];
    return p;
}
static LayerP readLayer(void* const* d, int& i) {
    LayerP L;
    L.b = readDir(d, i); L.f = readDir(d, i);
    L.in_proj_w = (const float*)d[i++]; L.ln_b = (const float*)d[i++];
    L.ln_w = (const float*)d[i++]; L.out_proj_w = (const float*)d[i++];
    L.s = readDir(d, i);
    return L;
}

static void run_mamba(const LayerP& P, Bufs& B, int Bn, int L, int dm, int ns, hipStream_t st)
{
    const int di = 2 * dm, E2 = 4 * dm;
    const int r  = (dm + 15) / 16;
    // in_proj: xz[E2,L] = W[E2,dm] * xf[L,dm]^T    (NT, dual-TDM pipelined)
    { dim3 g(cdiv(L, 64), cdiv(E2, 64), Bn);
      wmma_gemm_kernel<true><<<g, 128, 0, st>>>(P.in_proj_w, B.XF, B.XZ, E2, L, dm,
                                                0, (ll)L * dm, (ll)E2 * L); }
    const DirP* dirs[3] = { &P.f, &P.b, &P.s };
    for (int dd = 0; dd < 3; ++dd) {
        const int mode = dd;                    // 0=f, 1=b, 2=s
        const ll totE = (ll)Bn * E2 * L;
        const ll totD = (ll)Bn * di * L;
        const float* xzd = B.XZ;                // forward: no permute needed
        if (mode != 0) {
            seq_permute_kernel<<<gb(totE), 256, 0, st>>>(B.XZ, B.XZD, L, ns, mode, totE);
            xzd = B.XZD;
        }
        dwconv_silu_kernel<<<gb(totD), 256, 0, st>>>(xzd, dirs[dd]->conv_w,
                                                     dirs[dd]->conv_b, B.XC, E2, di, L, totD);
        // x_proj: xdbl[r+32,L] = Wx[r+32,di] * xc[di,L]   (NN)
        { dim3 g(cdiv(L, 64), cdiv(r + 32, 64), Bn);
          wmma_gemm_kernel<false><<<g, 128, 0, st>>>(dirs[dd]->xproj_w, B.XC, B.XDBL,
                                                     r + 32, L, di, 0, (ll)di * L, (ll)(r + 32) * L); }
        // dt_proj: delta[di,L] = Wd[di,r] * xdbl[:r,L]    (NN)
        { dim3 g(cdiv(L, 64), cdiv(di, 64), Bn);
          wmma_gemm_kernel<false><<<g, 128, 0, st>>>(dirs[dd]->dtproj_w, B.XDBL, B.DELTA,
                                                     di, L, r, 0, (ll)(r + 32) * L, (ll)di * L); }
        bias_softplus_kernel<<<gb(totD), 256, 0, st>>>(B.DELTA, dirs[dd]->dtproj_b, di, L, totD);
        { int nt = Bn * di;
          scan_kernel<<<cdiv(nt, 128), 128, 0, st>>>(B.DELTA, B.XC, B.XDBL, dirs[dd]->A_log,
                                                     dirs[dd]->D, xzd, B.YB, Bn, di, L, r, E2); }
        accum_out_kernel<<<gb(totD), 256, 0, st>>>(B.ACC, B.YB, L, ns, mode, dd == 0 ? 1 : 0, totD);
    }
    // out_proj: out[dm,L] = Wo[dm,di] * acc[di,L]   (NN)
    { dim3 g(cdiv(L, 64), cdiv(dm, 64), Bn);
      wmma_gemm_kernel<false><<<g, 128, 0, st>>>(P.out_proj_w, B.ACC, B.OUT, dm, L, di,
                                                 0, (ll)di * L, (ll)dm * L); }
}

static void tri_pixel_host(float* feat, const LayerP& P, Bufs& B, int C, int S, int p,
                           int ns, hipStream_t st)
{
    const int sp = S / p, L = sp * sp * sp, Bn = p * p * p;
    const ll tot = (ll)Bn * L * C;
    pix_gather_kernel<<<gb(tot), 256, 0, st>>>(feat, B.XF, C, S, p, tot);
    layernorm_kernel<<<gb((ll)Bn * L), 256, 0, st>>>(B.XF, P.ln_w, P.ln_b, (ll)Bn * L, C);
    run_mamba(P, B, Bn, L, C, ns, st);
    pix_scatter_kernel<<<gb(tot), 256, 0, st>>>(feat, B.OUT, C, S, p, tot);
}

static void tri_window_host(float* feat, const LayerP& P, Bufs& B, float* SMALL,
                            int C, int S, int p, int ns, hipStream_t st)
{
    const int sp = S / p, L = sp * sp * sp;
    const ll totS = (ll)C * L;
    avgpool_kernel<<<gb(totS), 256, 0, st>>>(feat, SMALL, C, S, p, totS);
    transpose_cl_lc_kernel<<<gb(totS), 256, 0, st>>>(SMALL, B.XF, C, L, totS);
    layernorm_kernel<<<gb((ll)L), 256, 0, st>>>(B.XF, P.ln_w, P.ln_b, (ll)L, C);
    run_mamba(P, B, 1, L, C, ns, st);
    const ll totX = (ll)C * S * S * S;
    trilinear_add_kernel<<<gb(totX), 256, 0, st>>>(feat, B.OUT, C, S, p, totX);
}

extern "C" void kernel_launch(void* const* d_in, const int* in_sizes, int n_in,
                              void* d_out, int out_size, void* d_ws, size_t ws_size,
                              hipStream_t stream)
{
    static const int DIMS[4]   = {48, 96, 192, 384};
    static const int NSL[4]    = {48, 24, 12, 6};
    static const int WINSZ[4]  = {8, 8, 4, 3};
    static const int SARR[4]   = {48, 24, 12, 6};

    // --- parameter pointer map (JAX pytree flatten: dict keys sorted) ---
    // d_in[0]=x; then down[0..2]{b,w}; stages[i][j]{pam(25), pim(25)}; stem_b; stem_w
    const float* x_in = (const float*)d_in[0];
    int ip = 1;
    const float *down_w[3], *down_b[3];
    for (int k = 0; k < 3; ++k) { down_b[k] = (const float*)d_in[ip++]; down_w[k] = (const float*)d_in[ip++]; }
    LayerP pam[4][2], pim[4][2];
    for (int s = 0; s < 4; ++s)
        for (int j = 0; j < 2; ++j) { pam[s][j] = readLayer(d_in, ip); pim[s][j] = readLayer(d_in, ip); }
    const float* stem_b = (const float*)d_in[ip++];
    const float* stem_w = (const float*)d_in[ip++];
    (void)in_sizes; (void)n_in; (void)out_size;

    // --- workspace carve (floats); peak set sized by stage-0 tri_pixel ---
    float* W = (float*)d_ws;
    ll off = 0;
    auto take = [&](ll n) { float* p = W + off; off += n; return p; };
    float* FEAT  = take(5308416);    // 48 * 48^3
    float* TMP   = take(5308416);
    float* SMALL = take(663552);     // 48 * 24^3
    Bufs B;
    B.XF    = take(5308416);         // 512*216*48
    B.XZ    = take(21233664);        // 512*192*216
    B.XZD   = take(21233664);
    B.XC    = take(10616832);
    B.XDBL  = take(3870720);         // 512*35*216
    B.DELTA = take(10616832);
    B.YB    = take(10616832);
    B.ACC   = take(10616832);
    B.OUT   = take(5308416);
    if ((size_t)off * sizeof(float) > ws_size) return;  // insufficient scratch

    float* outp = (float*)d_out;
    ll outOff[4]; { ll o = 0; for (int i = 0; i < 4; ++i) { outOff[i] = o; o += (ll)DIMS[i] * SARR[i] * SARR[i] * SARR[i]; } }

    float* featA = FEAT;
    float* featB = TMP;

    // stem: conv3d(1->48, k7 s2 p3) on 96^3, then instance norm
    conv3d_kernel<<<gb((ll)48 * 48 * 48 * 48), 256, 0, stream>>>(
        x_in, stem_w, stem_b, featB, 1, 96, 48, 48, 7, 2, 3, (ll)48 * 48 * 48 * 48);
    instnorm_kernel<<<48, 256, 0, stream>>>(featB, featA, (ll)48 * 48 * 48);

    for (int i = 0; i < 4; ++i) {
        const int C = DIMS[i];
        if (i > 0) {
            // conv3d(instance_norm(x)) — normalized prev-stage x lives in d_out segment
            const int Sp = SARR[i - 1], So = SARR[i];
            const ll tot = (ll)C * So * So * So;
            conv3d_kernel<<<gb(tot), 256, 0, stream>>>(
                outp + outOff[i - 1], down_w[i - 1], down_b[i - 1], featB,
                DIMS[i - 1], Sp, C, So, 2, 2, 0, tot);
            float* t2 = featA; featA = featB; featB = t2;
        }
        const int S = SARR[i];
        for (int j = 0; j < 2; ++j) {
            tri_pixel_host (featA, pim[i][j], B, C, S, WINSZ[i], NSL[i] / WINSZ[i], stream);
            tri_window_host(featA, pam[i][j], B, SMALL, C, S, 2, NSL[i] / 2, stream);
        }
        instnorm_kernel<<<C, 256, 0, stream>>>(featA, outp + outOff[i], (ll)S * S * S);
    }
}
